// GraphPool_79061757984936
// MI455X (gfx1250) — matrix operand order repaired and checked by reading, not compiled, not run on hardware
//
#include <hip/hip_runtime.h>

typedef __attribute__((ext_vector_type(16))) _Float16 v16h;
typedef __attribute__((ext_vector_type(8)))  _Float16 v8h;
typedef __attribute__((ext_vector_type(8)))  float    v8f;
typedef __attribute__((ext_vector_type(4)))  int      v4i;

#define N_NODES 16384
#define D_FEAT  64
#define KHALF   (N_NODES / 2)   // 8192 : K range per K-split wave

// ---------------------------------------------------------------------------
// Kernel 1: split fp32 x into hi/lo f16 halves, stored TRANSPOSED [D][N] so
// the WMMA B-fragment (K-major per lane) becomes two contiguous b128 loads.
// hi = f16(x); lo = f16(x - hi)  ->  ~22 effective mantissa bits.
// ---------------------------------------------------------------------------
__global__ __launch_bounds__(256)
void split_transpose_f16(const float* __restrict__ x,
                         _Float16* __restrict__ xt_hi,
                         _Float16* __restrict__ xt_lo) {
    int tid = blockIdx.x * blockDim.x + threadIdx.x;  // 0 .. N*D-1
    int d = tid >> 14;              // tid / 16384  -> feature index
    int n = tid & (N_NODES - 1);    // tid % 16384  -> node index
    float v = x[n * D_FEAT + d];
    _Float16 hi = (_Float16)v;
    _Float16 lo = (_Float16)(v - (float)hi);
    xt_hi[tid] = hi;                // tid == d*N + n : coalesced writes
    xt_lo[tid] = lo;
}

// adj entries are 0/1: f16(1.0)=0x3C00. (x | y<<16)*0x3C00 packs two f16
// exactly (x*0x3C00 < 2^16 -> no carry): 2 VALU per pair.
__device__ __forceinline__ unsigned pk(int lo, int hi) {
    return ((unsigned)lo | ((unsigned)hi << 16)) * 0x3C00u;
}

// ---------------------------------------------------------------------------
// Kernel 2: block = 64-row M-tile, 8 waves = (2 K-halves) x (4 strip-waves).
// Each wave: 16 rows x 8192 K of adj (read exactly once anywhere), all 4
// feature tiles. Per iteration: ALL loads (4 A + 16 B, distinct reg tuples)
// are issued before the 8 WMMAs -> clause-batched loads, progressive waits,
// loads of iter i+1 issue under WMMAs of iter i. K-halves reduce via LDS.
// ---------------------------------------------------------------------------
__global__ __launch_bounds__(256)
void graphpool_wmma(const int* __restrict__ adj,
                    const float* __restrict__ x,
                    const _Float16* __restrict__ xt_hi,
                    const _Float16* __restrict__ xt_lo,
                    float* __restrict__ out) {
    __shared__ float red[4 * 4 * 8 * 32];   // [ws][nn][v][lane] : 16 KiB

    const int blk  = blockIdx.x;            // 0..255 : 64-row M-tile
    const int wid  = threadIdx.x >> 5;      // 0..7
    const int ks   = wid >> 2;              // 0/1 : K-half
    const int ws   = wid & 3;               // 0..3 : strip in M-tile
    const int lane = threadIdx.x & 31;
    const int r    = lane & 15;
    const int h    = lane >> 4;

    // A-matrix: lane holds adj row; K-map per ISA 16-bit A layout.
    const int rowA = blk * 64 + ws * 16 + r;
    const int* __restrict__ ap = adj + (long)rowA * N_NODES + ks * KHALF;

    // B-matrix: lane holds feature column (nn*16 + r); K = kk + h*8 + e (e<8), kk + 16 + h*8 + (e-8) (e>=8).
    // nn handled via immediate offsets (nn*16*N halfs = 512 KiB < 24-bit imm).
    const long bofs = (long)r * N_NODES + (long)(ks * KHALF) + h * 8;
    const _Float16* __restrict__ bh = xt_hi + bofs;
    const _Float16* __restrict__ bl = xt_lo + bofs;

    v8f acc[4] = {{}, {}, {}, {}};

#pragma unroll 1
    for (int kk = 0; kk < KHALF; kk += 32) {
        // ---- issue A loads first (their wait comes earliest, for the pack)
        const int* a = ap + kk;
        v4i a0 = *(const v4i*)(a + h * 8);          // K = h*8 + 0..3
        v4i a1 = *(const v4i*)(a + h * 8 + 4);      // K = h*8 + 4..7
        v4i a2 = *(const v4i*)(a + 16 + h * 8);     // K = 16 + h*8 + 0..3
        v4i a3 = *(const v4i*)(a + 16 + h * 8 + 4); // K = 16 + h*8 + 4..7

        // ---- all 8 B fragments into DISTINCT tuples (live ranges overlap
        //      the whole WMMA block -> no register reuse, no serialization)
        v16h Bh[4], Bl[4];
#pragma unroll
        for (int nn = 0; nn < 4; ++nn) {
            const long no = (long)nn * (16 * N_NODES);
            union { v16h f; v8h p[2]; } th, tl;
            th.p[0] = *(const v8h*)(bh + no + kk);
            th.p[1] = *(const v8h*)(bh + no + kk + 16);
            tl.p[0] = *(const v8h*)(bl + no + kk);
            tl.p[1] = *(const v8h*)(bl + no + kk + 16);
            Bh[nn] = th.f;
            Bl[nn] = tl.f;
        }

        // ---- keep the HBM adj stream deep: prefetch ~16 iterations ahead
        __builtin_prefetch(ap + kk + 512, 0, 0);

        // ---- pack A: 16 ints -> 8 f16 pairs (exact), 2 VALU per pair
        union { v16h f; unsigned u[8]; } A;
        A.u[0] = pk(a0.x, a0.y);  A.u[1] = pk(a0.z, a0.w);
        A.u[2] = pk(a1.x, a1.y);  A.u[3] = pk(a1.z, a1.w);
        A.u[4] = pk(a2.x, a2.y);  A.u[5] = pk(a2.z, a2.w);
        A.u[6] = pk(a3.x, a3.y);  A.u[7] = pk(a3.z, a3.w);

        // ---- 8 WMMAs back-to-back; later B loads land under earlier WMMAs
#pragma unroll
        for (int nn = 0; nn < 4; ++nn) {
            acc[nn] = __builtin_amdgcn_wmma_f32_16x16x32_f16(
                          false, A.f, false, Bh[nn], (short)0, acc[nn],
                          false, false);
            acc[nn] = __builtin_amdgcn_wmma_f32_16x16x32_f16(
                          false, A.f, false, Bl[nn], (short)0, acc[nn],
                          false, false);
        }
    }

    // ---- K-split reduction through LDS (divergence only after all WMMAs)
    if (ks == 1) {
#pragma unroll
        for (int nn = 0; nn < 4; ++nn)
#pragma unroll
            for (int v = 0; v < 8; ++v)
                red[(((ws * 4 + nn) * 8 + v) << 5) | lane] = acc[nn][v];
    }
    __syncthreads();
    if (ks == 0) {
#pragma unroll
        for (int nn = 0; nn < 4; ++nn) {
#pragma unroll
            for (int v = 0; v < 8; ++v) {
                // C/D layout: VGPR v -> row v + 8h ; col = lane&15
                int  orow = blk * 64 + ws * 16 + v + h * 8;
                int  ocol = nn * 16 + r;
                long idx  = (long)orow * D_FEAT + ocol;
                float part = red[(((ws * 4 + nn) * 8 + v) << 5) | lane];
                out[idx] = acc[nn][v] + part + x[idx];  // + exact f32 self term
            }
        }
    }
}

// ---------------------------------------------------------------------------
extern "C" void kernel_launch(void* const* d_in, const int* in_sizes, int n_in,
                              void* d_out, int out_size, void* d_ws, size_t ws_size,
                              hipStream_t stream) {
    const float* x   = (const float*)d_in[0];   // [N, 64] fp32
    const int*   adj = (const int*)d_in[1];     // [N, N]  int32 (0/1)
    float*       out = (float*)d_out;           // [N, 64] fp32

    _Float16* xt_hi = (_Float16*)d_ws;                      // 2 MiB
    _Float16* xt_lo = xt_hi + (size_t)N_NODES * D_FEAT;     // 2 MiB

    split_transpose_f16<<<(N_NODES * D_FEAT) / 256, 256, 0, stream>>>(
        x, xt_hi, xt_lo);

    graphpool_wmma<<<N_NODES / 64, 256, 0, stream>>>(
        adj, x, xt_hi, xt_lo, out);
}
